// GraphAttentionNetwork_10514079941124
// MI455X (gfx1250) — compile-verified, matrix operand order checked by reading, not run on hardware
//
#include <hip/hip_runtime.h>
#include <hip/hip_bf16.h>

typedef __attribute__((ext_vector_type(16))) _Float16 v16h;
typedef __attribute__((ext_vector_type(8)))  _Float16 v8h;
typedef __attribute__((ext_vector_type(8)))  float    v8f;

#define NN   10000
#define MPAD 10048            // NN rounded up to 64 (GEMM tile height)
#define EE   320000
#define ETOT (EE + NN)

// ---------------------------------------------------------------------------
// CDNA5 async global->LDS copy (ASYNCcnt-tracked), 16 bytes per lane.
// LDS byte address = low 32 bits of the flat shared-memory pointer
// ---------------------------------------------------------------------------
__device__ __forceinline__ void async_copy_b128(unsigned lds_off, const void* gptr) {
    asm volatile("global_load_async_to_lds_b128 %0, %1, off"
                 :: "v"(lds_off), "v"(gptr) : "memory");
}
__device__ __forceinline__ void wait_async0() {
    asm volatile("s_wait_asynccnt 0x0" ::: "memory");
}

// ---------------------------------------------------------------------------
// WMMA GEMM: C[MPAD,Nc] = A16[MPAD,K] @ (Bt16[Nc,K])^T  (f16 in, f32 accum/out)
// Block: 256 threads (8 waves), tile 64x64, K-step 32.
// Wave w: M-subtile (w&3)*16, N-subtiles (w>>2)*32 + {0,16}.
// A16/C are MPAD-padded and Nc/K are compile-time multiples of 64/32:
// no guards anywhere -> pure async b128 copies + ds_load_b128 + wmma +
// immediate-offset stores.
// ---------------------------------------------------------------------------
template <int Nc, int K>
__global__ __launch_bounds__(256)
void gat_wmma_gemm(const _Float16* __restrict__ A, const _Float16* __restrict__ Bt,
                   float* __restrict__ C) {
    __shared__ __align__(16) _Float16 Al[64 * 40];  // [row][k], stride 40 halves
    __shared__ __align__(16) _Float16 Bl[64 * 40];  // [n][k]

    const int tid  = threadIdx.x;
    const int wave = tid >> 5;
    const int lane = tid & 31;
    const int half = lane >> 4;
    const int l16  = lane & 15;
    const int m0   = blockIdx.x * 64;
    const int n0   = blockIdx.y * 64;
    const int msub = (wave & 3) * 16;
    const int nsub = (wave >> 2) * 32;

    // Staging assignment: 256 threads x 16B = one 64x32-half tile per buffer.
    const int r  = tid >> 2;         // 0..63 (row of A tile / n of B tile)
    const int c8 = (tid & 3) * 8;    // half-element offset within the K slab
    const unsigned al_off = (unsigned)(uintptr_t)&Al[r * 40 + c8];
    const unsigned bl_off = (unsigned)(uintptr_t)&Bl[r * 40 + c8];
    const _Float16* ag = A  + (size_t)(m0 + r) * K + c8;
    const _Float16* bg = Bt + (size_t)(n0 + r) * K + c8;

    v8f c0 = {};
    v8f c1 = {};

    for (int k0 = 0; k0 < K; k0 += 32) {
        async_copy_b128(al_off, ag + k0);
        async_copy_b128(bl_off, bg + k0);
        wait_async0();
        __syncthreads();

        // Fragments: lane half (lane>>4) owns K = h*8..h*8+7 (elems 0-7)
        // and K = 16+h*8.. (elems 8-15); row/col = lane&15.
        v16h a, b0, b1;
        {
            const _Float16* p = &Al[(msub + l16) * 40 + half * 8];
            *(v8h*)&a         = *(const v8h*)p;
            *(((v8h*)&a) + 1) = *(const v8h*)(p + 16);
        }
        {
            const _Float16* p = &Bl[(nsub + l16) * 40 + half * 8];
            *(v8h*)&b0         = *(const v8h*)p;
            *(((v8h*)&b0) + 1) = *(const v8h*)(p + 16);
            p += 16 * 40;
            *(v8h*)&b1         = *(const v8h*)p;
            *(((v8h*)&b1) + 1) = *(const v8h*)(p + 16);
        }

        c0 = __builtin_amdgcn_wmma_f32_16x16x32_f16(false, a, false, b0,
                                                    (short)0, c0, false, false);
        c1 = __builtin_amdgcn_wmma_f32_16x16x32_f16(false, a, false, b1,
                                                    (short)0, c1, false, false);
        __syncthreads();
    }

    // Epilogue: acc VGPR rr holds M = msub + rr + 8*half, N = l16 (+16 for c1).
    // C is MPAD-padded -> unconditional stores with constant offsets.
    float* cp = C + (size_t)(m0 + msub + 8 * half) * Nc + (n0 + nsub + l16);
#pragma unroll
    for (int rr = 0; rr < 8; ++rr) {
        cp[rr * Nc]      = c0[rr];
        cp[rr * Nc + 16] = c1[rr];
    }
}

// ---------------------------------------------------------------------------
// Precision-conversion helpers (one-time, bandwidth-trivial)
// ---------------------------------------------------------------------------
// out[i] = f16(in[i]) for i < n_valid, 0 for n_valid <= i < n_total (row pad)
__global__ void gat_cvt_f16(const float* __restrict__ in, _Float16* __restrict__ out,
                            long n_valid, long n_total) {
    long i = (long)blockIdx.x * blockDim.x + threadIdx.x;
    if (i < n_total) out[i] = (i < n_valid) ? (_Float16)in[i] : (_Float16)0.0f;
}

// Wt[n*K + k] = f16(W[k*Nc + n])  (coalesced read over n)
__global__ void gat_cvt_f16_t(const float* __restrict__ W, _Float16* __restrict__ Wt,
                              int K, int Nc) {
    int idx = blockIdx.x * blockDim.x + threadIdx.x;
    if (idx >= K * Nc) return;
    int k = idx / Nc, n = idx % Nc;
    Wt[(size_t)n * K + k] = (_Float16)W[idx];
}

__global__ void gat_fill(float* __restrict__ p, float v, int n) {
    int i = blockIdx.x * blockDim.x + threadIdx.x;
    if (i < n) p[i] = v;
}

__global__ void gat_fill_f16(_Float16* __restrict__ p, long n) {
    long i = (long)blockIdx.x * blockDim.x + threadIdx.x;
    if (i < n) p[i] = (_Float16)0.0f;
}

// ---------------------------------------------------------------------------
// Per-node attention projections: as_n[n,h] = <h[n,h,:], a_src[h,:]>, same dst
// One wave32 per node; shuffle reduction.
// ---------------------------------------------------------------------------
__global__ __launch_bounds__(256)
void gat_node_dots(const float* __restrict__ h, const float* __restrict__ a_src,
                   const float* __restrict__ a_dst, float* __restrict__ as_n,
                   float* __restrict__ ad_n, int n_nodes, int heads, int F) {
    int wave = threadIdx.x >> 5, lane = threadIdx.x & 31;
    int node = blockIdx.x * 8 + wave;
    if (node >= n_nodes) return;
    for (int hh = 0; hh < heads; ++hh) {
        float s = 0.0f, d = 0.0f;
        const float* hp  = h + (size_t)node * heads * F + hh * F;
        const float* asp = a_src + hh * F;
        const float* adp = a_dst + hh * F;
        for (int f = lane; f < F; f += 32) {
            float v = hp[f];
            s += v * asp[f];
            d += v * adp[f];
        }
#pragma unroll
        for (int off = 16; off > 0; off >>= 1) {
            s += __shfl_xor(s, off, 32);
            d += __shfl_xor(d, off, 32);
        }
        if (lane == 0) {
            as_n[node * heads + hh] = s;
            ad_n[node * heads + hh] = d;
        }
    }
}

__device__ __forceinline__ void edge_sd(const int* __restrict__ ei, int e,
                                        int& s, int& d) {
    if (e < EE) { s = ei[e]; d = ei[EE + e]; }
    else        { s = d = e - EE; }  // appended self-loops
}

__device__ __forceinline__ float leaky(float x) {
    return x > 0.0f ? x : 0.2f * x;
}

// Orderable-int trick -> native GLOBAL_ATOMIC_MAX_I32 / MIN_U32
__device__ __forceinline__ void atomicMaxF(float* addr, float val) {
    if (val >= 0.0f) atomicMax((int*)addr, __float_as_int(val));
    else             atomicMin((unsigned int*)addr, (unsigned int)__float_as_int(val));
}

__global__ void gat_edge_max(const int* __restrict__ ei,
                             const float* __restrict__ as_n,
                             const float* __restrict__ ad_n,
                             float* __restrict__ mx, int heads) {
    int e = blockIdx.x * blockDim.x + threadIdx.x;
    if (e >= ETOT) return;
    int s, d; edge_sd(ei, e, s, d);
    for (int hh = 0; hh < heads; ++hh) {
        float lg = leaky(as_n[s * heads + hh] + ad_n[d * heads + hh]);
        atomicMaxF(&mx[d * heads + hh], lg);
    }
}

__global__ void gat_edge_expsum(const int* __restrict__ ei,
                                const float* __restrict__ as_n,
                                const float* __restrict__ ad_n,
                                const float* __restrict__ mx,
                                float* __restrict__ sm,
                                float* __restrict__ ex, int heads) {
    int e = blockIdx.x * blockDim.x + threadIdx.x;
    if (e >= ETOT) return;
    int s, d; edge_sd(ei, e, s, d);
    for (int hh = 0; hh < heads; ++hh) {
        float lg = leaky(as_n[s * heads + hh] + ad_n[d * heads + hh]);
        float v  = __expf(lg - mx[d * heads + hh]);
        ex[(size_t)e * heads + hh] = v;
        unsafeAtomicAdd(&sm[d * heads + hh], v);  // native global_atomic_add_f32
    }
}

// One wave32 per edge: agg[dst,h,:] += h[src,h,:] * (ex / segsum)
__global__ __launch_bounds__(256)
void gat_edge_agg(const int* __restrict__ ei, const float* __restrict__ h,
                  const float* __restrict__ ex, const float* __restrict__ sm,
                  float* __restrict__ agg, int heads, int F) {
    int wave = threadIdx.x >> 5, lane = threadIdx.x & 31;
    int e = blockIdx.x * 8 + wave;
    if (e >= ETOT) return;
    int s, d; edge_sd(ei, e, s, d);
    for (int hh = 0; hh < heads; ++hh) {
        float w = ex[(size_t)e * heads + hh] / sm[d * heads + hh];
        const float* hp = h   + (size_t)s * heads * F + hh * F;
        float*       op = agg + (size_t)d * heads * F + hh * F;
        for (int f = lane; f < F; f += 32)
            unsafeAtomicAdd(&op[f], hp[f] * w);
    }
}

// Layer-1 epilogue: ELU(agg + b) -> f16 directly into padded layer-2 A16.
__global__ void gat_bias_elu_f16(const float* __restrict__ agg,
                                 const float* __restrict__ b,
                                 _Float16* __restrict__ out16,
                                 int n_nodes, int dim) {
    int i = blockIdx.x * blockDim.x + threadIdx.x;
    if (i >= n_nodes * dim) return;
    float v = agg[i] + b[i % dim];
    v = v > 0.0f ? v : (__expf(v) - 1.0f);
    out16[i] = (_Float16)v;
}

// Layer-2 epilogue: out = agg + b
__global__ void gat_bias_add(float* __restrict__ p, const float* __restrict__ b,
                             int n_nodes, int dim) {
    int i = blockIdx.x * blockDim.x + threadIdx.x;
    if (i >= n_nodes * dim) return;
    p[i] = p[i] + b[i % dim];
}

// ---------------------------------------------------------------------------
extern "C" void kernel_launch(void* const* d_in, const int* in_sizes, int n_in,
                              void* d_out, int out_size, void* d_ws, size_t ws_size,
                              hipStream_t stream) {
    const float* x   = (const float*)d_in[0];
    const int*   ei  = (const int*)d_in[1];
    const float* W1  = (const float*)d_in[2];
    const float* a1s = (const float*)d_in[3];
    const float* a1d = (const float*)d_in[4];
    const float* b1  = (const float*)d_in[5];
    const float* W2  = (const float*)d_in[6];
    const float* a2s = (const float*)d_in[7];
    const float* a2d = (const float*)d_in[8];
    const float* b2  = (const float*)d_in[9];
    float*       out = (float*)d_out;

    // ---- Workspace layout: f32 region, then f16 region (all 16B aligned) ----
    float* ws   = (float*)d_ws;
    float* h1   = ws;                          // MPAD*512 (GEMM1 out, padded)
    float* agg1 = h1   + (size_t)MPAD * 512;   // NN*512
    float* as1  = agg1 + (size_t)NN * 512;     // NN*4
    float* ad1  = as1  + NN * 4;
    float* mx1  = ad1  + NN * 4;
    float* sm1  = mx1  + NN * 4;
    float* ex1  = sm1  + NN * 4;               // ETOT*4
    float* h2   = ex1  + (size_t)ETOT * 4;     // MPAD*256 (GEMM2 out, padded)
    float* as2  = h2   + (size_t)MPAD * 256;   // NN
    float* ad2  = as2  + NN;
    float* mx2  = ad2  + NN;
    float* sm2  = mx2  + NN;
    float* ex2  = sm2  + NN;                   // ETOT

    _Float16* x16 = (_Float16*)(ex2 + ETOT);   // MPAD*256 (pad rows = 0)
    _Float16* W1t = x16 + (size_t)MPAD * 256;  // 512*256  [n][k]
    _Float16* a16 = W1t + (size_t)512 * 256;   // MPAD*512 (ELU out, pad rows = 0)
    _Float16* W2t = a16 + (size_t)MPAD * 512;  // 256*512  [n][k]

    const dim3 blk(256);
    const int nodeBlocks     = (NN + 7) / 8;
    const int edgeBlocks     = (ETOT + 255) / 256;
    const int edgeWaveBlocks = (ETOT + 7) / 8;

    // ---- One-time f16 conversions (also zero the pad rows) ----
    gat_cvt_f16  <<<(int)(((long)MPAD * 256 + 255) / 256), blk, 0, stream>>>(
        x, x16, (long)NN * 256, (long)MPAD * 256);
    gat_cvt_f16_t<<<(512 * 256 + 255) / 256, blk, 0, stream>>>(W1, W1t, 256, 512);
    gat_cvt_f16_t<<<(512 * 256 + 255) / 256, blk, 0, stream>>>(W2, W2t, 512, 256);
    gat_fill_f16 <<<(int)(((long)MPAD * 512 + 255) / 256), blk, 0, stream>>>(
        a16, (long)MPAD * 512);  // valid rows overwritten after layer 1

    // ---------------- Layer 1: in=256, heads=4, hid=128, concat ----------------
    gat_wmma_gemm<512, 256><<<dim3(MPAD / 64, 512 / 64), blk, 0, stream>>>(x16, W1t, h1);
    gat_node_dots<<<nodeBlocks, blk, 0, stream>>>(h1, a1s, a1d, as1, ad1, NN, 4, 128);

    gat_fill<<<(NN * 4 + 255) / 256, blk, 0, stream>>>(mx1, -3.0e38f, NN * 4);
    gat_fill<<<(NN * 4 + 255) / 256, blk, 0, stream>>>(sm1, 0.0f, NN * 4);
    gat_fill<<<(NN * 512 + 255) / 256, blk, 0, stream>>>(agg1, 0.0f, NN * 512);

    gat_edge_max   <<<edgeBlocks, blk, 0, stream>>>(ei, as1, ad1, mx1, 4);
    gat_edge_expsum<<<edgeBlocks, blk, 0, stream>>>(ei, as1, ad1, mx1, sm1, ex1, 4);
    gat_edge_agg   <<<edgeWaveBlocks, blk, 0, stream>>>(ei, h1, ex1, sm1, agg1, 4, 128);
    gat_bias_elu_f16<<<(NN * 512 + 255) / 256, blk, 0, stream>>>(agg1, b1, a16, NN, 512);

    // ---------------- Layer 2: in=512, heads=1, out=256 ------------------------
    gat_wmma_gemm<256, 512><<<dim3(MPAD / 64, 256 / 64), blk, 0, stream>>>(a16, W2t, h2);
    gat_node_dots<<<nodeBlocks, blk, 0, stream>>>(h2, a2s, a2d, as2, ad2, NN, 1, 256);

    gat_fill<<<(NN + 255) / 256, blk, 0, stream>>>(mx2, -3.0e38f, NN);
    gat_fill<<<(NN + 255) / 256, blk, 0, stream>>>(sm2, 0.0f, NN);
    gat_fill<<<(NN * 256 + 255) / 256, blk, 0, stream>>>(out, 0.0f, NN * 256);

    gat_edge_max   <<<edgeBlocks, blk, 0, stream>>>(ei, as2, ad2, mx2, 1);
    gat_edge_expsum<<<edgeBlocks, blk, 0, stream>>>(ei, as2, ad2, mx2, sm2, ex2, 1);
    gat_edge_agg   <<<edgeWaveBlocks, blk, 0, stream>>>(ei, h2, ex2, sm2, out, 1, 256);
    gat_bias_add   <<<(NN * 256 + 255) / 256, blk, 0, stream>>>(out, b2, NN, 256);
}
